// global_conv_1219770712073
// MI455X (gfx1250) — matrix operand order repaired and checked by reading, not compiled, not run on hardware
//
#include <hip/hip_runtime.h>
#include <hip/hip_bf16.h>

typedef __bf16 v16bf __attribute__((ext_vector_type(16)));
typedef float  v8f   __attribute__((ext_vector_type(8)));
typedef unsigned int uint32;

// ---------------------------------------------------------------------------
// CDNA5 WMMA helpers (wave32). Operands use contiguous k order: k = 16*(lane/16) + slot.
// ---------------------------------------------------------------------------
__device__ __forceinline__ v8f wmma_bf16(v16bf a, v16bf b, v8f c) {
  // D = A(16x32 bf16) x B(32x16 bf16) + C(16x16 f32)
  return __builtin_amdgcn_wmma_f32_16x16x32_bf16(false, a, false, b, (short)0, c,
                                                 false, false);
}

// A fragment: 16x32 bf16 tile, row-major in LDS with row stride `ld`.
// lanes 0-15: row=lane,    K = {0..15}
// lanes 16-31: row=lane-16, K = {16..31}
// (contiguous 8-element runs -> lowers to ds_load_2addr_b32 pairs)
__device__ __forceinline__ v16bf frag_a(const __bf16* p, int ld) {
  int lane = threadIdx.x & 31;
  int row  = lane & 15;
  int kb   = (lane >> 4) << 4;   // 0 or 16
  const __bf16* r = p + row * ld;
  v16bf a;
#pragma unroll
  for (int i = 0; i < 8; ++i) a[i] = r[kb + i];
#pragma unroll
  for (int i = 0; i < 8; ++i) a[8 + i] = r[kb + 8 + i];
  return a;
}

// Fragment-ready B layout: Bsw[ntile][lane][16] so each lane reads its 16 bf16
// contiguously (2x ds_load_b128).  Element (k 0..31, col 0..15) lives at
// lane = col | (((k>>4)&1)<<4), slot i = k&15.
__device__ __forceinline__ int b_lane(int k, int col) {
  return (col & 15) | (((k >> 4) & 1) << 4);
}

__device__ __forceinline__ uint32 pk_f32_bf16(float a, float b) {
  union { __bf16 h[2]; uint32 u; } v;
  v.h[0] = (__bf16)a; v.h[1] = (__bf16)b;
  return v.u;
}
__device__ __forceinline__ uint32 pk_bf16x2(__bf16 a, __bf16 b) {
  union { __bf16 h[2]; uint32 u; } v;
  v.h[0] = a; v.h[1] = b;
  return v.u;
}

// ---------------------------------------------------------------------------
// Kernel 1: img_embed[n][1024][256] (bf16) = w_img[1024x2048] x img[n][2048][256]
// grid (4 Ntiles, 8 Mtiles, 64 batches), 256 threads (8 waves).
// Bandwidth-dominated (134 MB image stream): register double-buffer pipeline
// hides global latency under the WMMAs; all LDS commits are packed dwords;
// output tile staged in LDS for coalesced dword stores.
// ---------------------------------------------------------------------------
__global__ __launch_bounds__(256) void k_img_embed(
    const float* __restrict__ img, const float* __restrict__ w,
    __bf16* __restrict__ embed) {
  __shared__ __align__(32) union {
    struct { __bf16 Bsw[4][32][16]; __bf16 As[128][34]; } g;
    __bf16 Ot[128][64];
  } sh;
  const int tid = threadIdx.x;
  const int batch = blockIdx.z;
  const int rowBase = blockIdx.y * 128;
  const int colBase = blockIdx.x * 64;
  const int wv = tid >> 5;
  const int lane = tid & 31;
  const size_t imgBase = (size_t)batch * 2048 * 256;

  v8f acc[4] = {};
  float2 ra[8];
  float rb[8];
  // prologue: prefetch K-slice 0 (A: k-pairs as float2 -> global_load_b64)
#pragma unroll
  for (int e = 0; e < 8; ++e) {
    int f = tid + e * 256, r = f >> 4, c = (f & 15) * 2;
    ra[e] = *(const float2*)&w[(size_t)(rowBase + r) * 2048 + c];
  }
#pragma unroll
  for (int e = 0; e < 4; ++e) {
    int f = tid + e * 256, kp = f >> 6, col = f & 63, k = kp * 2;
    rb[2 * e]     = img[imgBase + (size_t)k * 256 + colBase + col];
    rb[2 * e + 1] = img[imgBase + (size_t)(k + 1) * 256 + colBase + col];
  }

  for (int kk = 0; kk < 2048; kk += 32) {
    // commit prefetched regs -> LDS as packed bf16 dwords
#pragma unroll
    for (int e = 0; e < 8; ++e) {
      int f = tid + e * 256, r = f >> 4, c = (f & 15) * 2;
      *(uint32*)&sh.g.As[r][c] = pk_f32_bf16(ra[e].x, ra[e].y);
    }
#pragma unroll
    for (int e = 0; e < 4; ++e) {
      int f = tid + e * 256, kp = f >> 6, col = f & 63, k = kp * 2;
      *(uint32*)&sh.g.Bsw[col >> 4][b_lane(k, col)][k & 15] =
          pk_f32_bf16(rb[2 * e], rb[2 * e + 1]);
    }
    __syncthreads();
    // prefetch next K-slice while WMMAs run
    int kn = kk + 32;
    if (kn < 2048) {
#pragma unroll
      for (int e = 0; e < 8; ++e) {
        int f = tid + e * 256, r = f >> 4, c = (f & 15) * 2;
        ra[e] = *(const float2*)&w[(size_t)(rowBase + r) * 2048 + kn + c];
      }
#pragma unroll
      for (int e = 0; e < 4; ++e) {
        int f = tid + e * 256, kp = f >> 6, col = f & 63, k = kp * 2;
        rb[2 * e]     = img[imgBase + (size_t)(kn + k) * 256 + colBase + col];
        rb[2 * e + 1] = img[imgBase + (size_t)(kn + k + 1) * 256 + colBase + col];
      }
    }
    v16bf a = frag_a(&sh.g.As[wv * 16][0], 34);
#pragma unroll
    for (int nt = 0; nt < 4; ++nt) {
      v16bf b = *(const v16bf*)&sh.g.Bsw[nt][lane][0];
      acc[nt] = wmma_bf16(a, b, acc[nt]);
    }
    __syncthreads();
  }
  // epilogue: stage tile in LDS (overlay), then coalesced dword stores
  int col = lane & 15, rbi = (lane >> 4) << 3;
#pragma unroll
  for (int nt = 0; nt < 4; ++nt)
#pragma unroll
    for (int r = 0; r < 8; ++r)
      sh.Ot[wv * 16 + rbi + r][nt * 16 + col] = (__bf16)acc[nt][r];
  __syncthreads();
  uint32* dstU =
      (uint32*)(embed + (size_t)batch * 1024 * 256 + (size_t)rowBase * 256 + colBase);
#pragma unroll
  for (int e = 0; e < 16; ++e) {
    int f = tid + e * 256, r = f >> 5, c2 = f & 31;
    dstU[(size_t)r * 128 + c2] = *(const uint32*)&sh.Ot[r][c2 * 2];
  }
}

// ---------------------------------------------------------------------------
// Kernel 2: scores = text[n][64][1024] x embed[n][1024][256]; fused softmax/T.
// One block per batch; 64 KB LDS union (GEMM staging | scores).
// ---------------------------------------------------------------------------
__global__ __launch_bounds__(256) void k_attn(
    const float* __restrict__ text, const __bf16* __restrict__ embed,
    float* __restrict__ attn_out, __bf16* __restrict__ attn_bf) {
  __shared__ __align__(32) union {
    struct { __bf16 Bsw[16][32][16]; __bf16 As[64][34]; } g;
    float Sc[64][256];
  } sh;
  const int tid = threadIdx.x;
  const int batch = blockIdx.x;
  const int wv = tid >> 5;
  const int mt = wv >> 1;          // M tile 0..3 (rows 16*mt)
  const int nh = wv & 1;           // N half: tiles 8*nh..
  const int lane = tid & 31;
  const size_t tBase = (size_t)batch * 64 * 1024;
  const size_t eBase = (size_t)batch * 1024 * 256;

  v8f acc[8] = {};
  for (int kk = 0; kk < 1024; kk += 32) {
#pragma unroll
    for (int e = 0; e < 4; ++e) {            // A: 64x32, k-pair float2 loads
      int f = tid + e * 256, r = f >> 4, c = (f & 15) * 2;
      float2 v = *(const float2*)&text[tBase + (size_t)r * 1024 + kk + c];
      *(uint32*)&sh.g.As[r][c] = pk_f32_bf16(v.x, v.y);
    }
#pragma unroll
    for (int e = 0; e < 16; ++e) {           // B: 32x256 bf16, k-pairs -> dwords
      int f = tid + e * 256;                 // 4096 pairs
      int kp = f >> 8, c = f & 255, k = kp * 2;
      __bf16 v0 = embed[eBase + (size_t)(kk + k) * 256 + c];
      __bf16 v1 = embed[eBase + (size_t)(kk + k + 1) * 256 + c];
      *(uint32*)&sh.g.Bsw[c >> 4][b_lane(k, c)][k & 15] = pk_bf16x2(v0, v1);
    }
    __syncthreads();
    v16bf a = frag_a(&sh.g.As[mt * 16][0], 34);
#pragma unroll
    for (int nt = 0; nt < 8; ++nt) {
      v16bf b = *(const v16bf*)&sh.g.Bsw[nh * 8 + nt][lane][0];
      acc[nt] = wmma_bf16(a, b, acc[nt]);
    }
    __syncthreads();
  }
  // dump scores (overlays GEMM staging; last barrier already passed)
  int col = lane & 15, rbs = (lane >> 4) << 3;
#pragma unroll
  for (int nt = 0; nt < 8; ++nt)
#pragma unroll
    for (int r = 0; r < 8; ++r)
      sh.Sc[mt * 16 + rbs + r][nh * 128 + nt * 16 + col] = acc[nt][r];
  __syncthreads();

  // softmax(scores/4): 4 lanes per row, wave32 shuffle reduction
  {
    int row = tid >> 2, part = tid & 3;
    int j0 = part * 64;
    float m = -3.4e38f;
    for (int j = 0; j < 64; ++j) m = fmaxf(m, sh.Sc[row][j0 + j]);
    m = fmaxf(m, __shfl_xor(m, 1, 32));
    m = fmaxf(m, __shfl_xor(m, 2, 32));
    float s = 0.f;
    for (int j = 0; j < 64; ++j) {
      float e = __expf((sh.Sc[row][j0 + j] - m) * 0.25f);
      sh.Sc[row][j0 + j] = e;
      s += e;
    }
    s += __shfl_xor(s, 1, 32);
    s += __shfl_xor(s, 2, 32);
    float inv = 1.f / s;
    size_t o = (size_t)batch * 64 * 256 + (size_t)row * 256 + j0;
    for (int j = 0; j < 64; j += 2) {
      float p0 = sh.Sc[row][j0 + j] * inv;
      float p1 = sh.Sc[row][j0 + j + 1] * inv;
      attn_out[o + j] = p0;
      attn_out[o + j + 1] = p1;
      *(uint32*)&attn_bf[o + j] = pk_f32_bf16(p0, p1);   // packed bf16 store
    }
  }
}

// ---------------------------------------------------------------------------
// Kernel 3: x[n][0:1024][64] = text^T (fp32 -> bf16, LDS-tiled transpose so
// both the global reads and writes are coalesced).
// grid (16 ctiles, 64 batches); tile = 64 l x 64 c.
// ---------------------------------------------------------------------------
__global__ __launch_bounds__(256) void k_text_x(const float* __restrict__ text,
                                                __bf16* __restrict__ x) {
  __shared__ __bf16 T[64][65];
  const int tid = threadIdx.x;
  const int n = blockIdx.y;
  const int c0 = blockIdx.x * 64;
  const float* src = text + (size_t)n * 64 * 1024 + c0;
#pragma unroll
  for (int e = 0; e < 16; ++e) {             // read text[l][c0+c], coalesced
    int f = tid + e * 256;                   // 4096
    int l = f >> 6, c = f & 63;
    T[l][c] = (__bf16)src[(size_t)l * 1024 + c];
  }
  __syncthreads();
  __bf16* dst = x + (size_t)n * 2048 * 64 + (size_t)c0 * 64;
#pragma unroll
  for (int e = 0; e < 16; ++e) {             // write x[c][l], coalesced
    int f = tid + e * 256;
    int c = f >> 6, l = f & 63;
    dst[(size_t)c * 64 + l] = T[l][c];
  }
}

// ---------------------------------------------------------------------------
// Kernel 4: img_cat[n][1024][64] = embed[n][1024][256] x attn[n]^T[256][64]
// also writes x[n][1024+c][64] (bf16 concat half). grid (8 Mtiles, 64 batch).
// attn^T tiles fill the swizzled B layout with straight dword copies (source
// is contiguous along k).
// ---------------------------------------------------------------------------
__global__ __launch_bounds__(256) void k_img_cat(
    const __bf16* __restrict__ embed, const __bf16* __restrict__ attn_bf,
    float* __restrict__ img_cat, __bf16* __restrict__ x) {
  __shared__ __align__(32) __bf16 Bsw[4][32][16];
  __shared__ __bf16 As[128][34];
  const int tid = threadIdx.x;
  const int batch = blockIdx.y;
  const int rowBase = blockIdx.x * 128;
  const int wv = tid >> 5;
  const int lane = tid & 31;
  const size_t eBase = (size_t)batch * 1024 * 256;
  const size_t aBase = (size_t)batch * 64 * 256;

  v8f acc[4] = {};
  for (int kk = 0; kk < 256; kk += 32) {
    const uint32* srcA = (const uint32*)(embed + eBase + (size_t)rowBase * 256 + kk);
#pragma unroll
    for (int e = 0; e < 8; ++e) {            // A: 128x32 bf16 copy (packed)
      int f = tid + e * 256;                 // 2048 dwords
      int r = f >> 4, c2 = f & 15;
      *(uint32*)&As[r][c2 * 2] = srcA[(size_t)r * 128 + c2];
    }
#pragma unroll
    for (int e = 0; e < 4; ++e) {            // B[k=s][l] = attn[l][kk+k]
      int f = tid + e * 256;                 // 1024 k-pairs
      int kp = f & 15, c = f >> 4, k = kp * 2;
      uint32 u = *(const uint32*)&attn_bf[aBase + (size_t)c * 256 + kk + k];
      *(uint32*)&Bsw[c >> 4][b_lane(k, c)][k & 15] = u;
    }
    __syncthreads();
    v16bf a = frag_a(&As[wv * 16][0], 34);
#pragma unroll
    for (int nt = 0; nt < 4; ++nt) {
      v16bf b = *(const v16bf*)&Bsw[nt][lane][0];
      acc[nt] = wmma_bf16(a, b, acc[nt]);
    }
    __syncthreads();
  }
  int col = lane & 15, rbi = (lane >> 4) << 3;
#pragma unroll
  for (int nt = 0; nt < 4; ++nt)
#pragma unroll
    for (int r = 0; r < 8; ++r) {
      int row = rowBase + wv * 16 + rbi + r;  // channel 0..1023
      int l = nt * 16 + col;
      float v = acc[nt][r];
      img_cat[(size_t)batch * 65536 + (size_t)row * 64 + l] = v;
      x[(size_t)batch * 131072 + (size_t)(1024 + row) * 64 + l] = (__bf16)v;
    }
}

// ---------------------------------------------------------------------------
// Kernel 5: inception convs (k=1 for co<512, k=3 as 3 shifted-B passes) for
// both gamma and beta (shared B tiles, 2 WMMAs per fragment), fused with the
// final gamma*img_cat+beta epilogue. grid (8 co-tiles, 64 batches).
// ---------------------------------------------------------------------------
__global__ __launch_bounds__(256) void k_incept(
    const __bf16* __restrict__ x, const float* __restrict__ img_cat,
    const float* __restrict__ wg1, const float* __restrict__ bg1,
    const float* __restrict__ wg3, const float* __restrict__ bg3,
    const float* __restrict__ wb1, const float* __restrict__ bb1,
    const float* __restrict__ wb3, const float* __restrict__ bb3,
    float* __restrict__ out) {
  __shared__ __align__(32) __bf16 Bsw[4][32][16];
  __shared__ __bf16 Ag[128][34];
  __shared__ __bf16 Ab[128][34];
  const int tid = threadIdx.x;
  const int batch = blockIdx.y;
  const int bx = blockIdx.x;                 // 0-3: k=1 branch, 4-7: k=3 branch
  const bool k3 = bx >= 4;
  const int coBase = bx * 128;
  const int wBase = k3 ? (coBase - 512) : coBase;
  const int wv = tid >> 5;
  const int lane = tid & 31;
  const size_t xBase = (size_t)batch * 131072;

  v8f accg[4] = {}, accb[4] = {};
  const int tlo = k3 ? 0 : 1, thi = k3 ? 3 : 2;
  for (int t = tlo; t < thi; ++t) {
    const int shift = t - 1;
    for (int kk = 0; kk < 2048; kk += 32) {
#pragma unroll
      for (int e = 0; e < 8; ++e) {          // gamma + beta weights, k-pairs
        int f = tid + e * 256, r = f >> 4, c = (f & 15) * 2;
        size_t wi = (size_t)(wBase + r) * 2048 + kk + c;
        if (k3) {
          *(uint32*)&Ag[r][c] = pk_f32_bf16(wg3[wi * 3 + t], wg3[(wi + 1) * 3 + t]);
          *(uint32*)&Ab[r][c] = pk_f32_bf16(wb3[wi * 3 + t], wb3[(wi + 1) * 3 + t]);
        } else {
          float2 g2 = *(const float2*)&wg1[wi];
          float2 b2 = *(const float2*)&wb1[wi];
          *(uint32*)&Ag[r][c] = pk_f32_bf16(g2.x, g2.y);
          *(uint32*)&Ab[r][c] = pk_f32_bf16(b2.x, b2.y);
        }
      }
#pragma unroll
      for (int e = 0; e < 8; ++e) {          // B: shifted x columns, zero pad
        int f = tid + e * 256;
        int k = f >> 6, c = f & 63;
        int lc = c + shift;
        __bf16 v = (__bf16)0.f;
        if (lc >= 0 && lc < 64) v = x[xBase + (size_t)(kk + k) * 64 + lc];
        Bsw[c >> 4][b_lane(k, c)][k & 15] = v;
      }
      __syncthreads();
      v16bf ag = frag_a(&Ag[wv * 16][0], 34);
      v16bf ab = frag_a(&Ab[wv * 16][0], 34);
#pragma unroll
      for (int nt = 0; nt < 4; ++nt) {
        v16bf b = *(const v16bf*)&Bsw[nt][lane][0];
        accg[nt] = wmma_bf16(ag, b, accg[nt]);
        accb[nt] = wmma_bf16(ab, b, accb[nt]);
      }
      __syncthreads();
    }
  }
  int col = lane & 15, rbi = (lane >> 4) << 3;
#pragma unroll
  for (int nt = 0; nt < 4; ++nt)
#pragma unroll
    for (int r = 0; r < 8; ++r) {
      int co = coBase + wv * 16 + rbi + r;
      int l = nt * 16 + col;
      int wco = k3 ? (co - 512) : co;
      float g = accg[nt][r] + (k3 ? bg3[wco] : bg1[wco]);
      float b = accb[nt][r] + (k3 ? bb3[wco] : bb1[wco]);
      size_t o = (size_t)batch * 65536 + (size_t)co * 64 + l;
      out[o] = g * img_cat[o] + b;
    }
}

// ---------------------------------------------------------------------------
extern "C" void kernel_launch(void* const* d_in, const int* in_sizes, int n_in,
                              void* d_out, int out_size, void* d_ws,
                              size_t ws_size, hipStream_t stream) {
  const float* img   = (const float*)d_in[0];   // [64][2048][16][16]
  const float* text  = (const float*)d_in[1];   // [64][64][1024]
  const float* w_img = (const float*)d_in[2];   // [1024][2048]
  const float* wg1   = (const float*)d_in[3];
  const float* bg1   = (const float*)d_in[4];
  const float* wg3   = (const float*)d_in[5];
  const float* bg3   = (const float*)d_in[6];
  const float* wb1   = (const float*)d_in[7];
  const float* bb1   = (const float*)d_in[8];
  const float* wb3   = (const float*)d_in[9];
  const float* bb3   = (const float*)d_in[10];

  float* out_main = (float*)d_out;                        // [64][1024][64]
  float* out_attn = out_main + (size_t)64 * 1024 * 64;    // [64][64][256]

  char* ws = (char*)d_ws;
  __bf16* embed_bf = (__bf16*)ws;                          // 32 MB
  size_t off = (size_t)64 * 1024 * 256 * 2;
  __bf16* attn_bf = (__bf16*)(ws + off);                   // 2 MB
  off += (size_t)64 * 64 * 256 * 2;
  float* img_cat = (float*)(ws + off);                     // 16 MB
  off += (size_t)64 * 1024 * 64 * 4;
  __bf16* x_bf = (__bf16*)(ws + off);                      // 16 MB

  k_img_embed<<<dim3(4, 8, 64), 256, 0, stream>>>(img, w_img, embed_bf);
  k_attn<<<dim3(64), 256, 0, stream>>>(text, embed_bf, out_attn, attn_bf);
  k_text_x<<<dim3(16, 64), 256, 0, stream>>>(text, x_bf);
  k_img_cat<<<dim3(8, 64), 256, 0, stream>>>(embed_bf, attn_bf, img_cat, x_bf);
  k_incept<<<dim3(8, 64), 256, 0, stream>>>(x_bf, img_cat, wg1, bg1, wg3, bg3,
                                            wb1, bb1, wb3, bb3, out_main);
}
